// Model_36601711297140
// MI455X (gfx1250) — compile-verified, matrix operand order checked by reading, not run on hardware
//
#include <hip/hip_runtime.h>
#include <math.h>

typedef __attribute__((ext_vector_type(16))) _Float16 v16h;
typedef __attribute__((ext_vector_type(8)))  float    v8f;

constexpr int kB = 8, kT = 8, kN = 1500, kCODE = 48, kG = 32, kH = 150;
constexpr int kD = 768, kHEADS = 4, kHD = 192, kEMB = 128, kFF = 3072, kL = 128, kXA = 64;
constexpr float kNEG = -1.0e9f;

__device__ __forceinline__ float sigmoidf_(float x) { return 1.f / (1.f + __expf(-x)); }

// Issue one per-lane 16B async global->LDS copy (gfx1250, tracked by ASYNCcnt).
__device__ __forceinline__ void async_g2l_b128(unsigned ldsOff, const float* gaddr)
{
  asm volatile("global_load_async_to_lds_b128 %0, %1, off"
               :: "v"(ldsOff), "v"(gaddr) : "memory");
}
__device__ __forceinline__ void wait_asynccnt0()
{
  asm volatile("s_wait_asynccnt 0" ::: "memory");
}

// ---------------------------------------------------------------------------
// Generic WMMA GEMM: C = act(alpha * (A @ B[^T]) + bias), strided batch.
// Block: 128 threads = 4 wave32s; 64x64 block tile; each wave owns a 32x32
// quadrant = 2x2 accumulators -> 4 x v_wmma_f32_16x16x32_f16 per K-step.
// Fast path (full tile, aligned): global_load_async_to_lds_b128 staging of
// f32 tiles, s_wait_asynccnt 0, then LDS f32 -> f16 conversion.
// ---------------------------------------------------------------------------
__global__ __launch_bounds__(128)
void gemm_wmma(const float* __restrict__ A, const float* __restrict__ Bm,
               const float* __restrict__ bias, float* __restrict__ C,
               int M, int N, int K, int lda, int ldb, int ldc,
               long long sA, long long sB, long long sC,
               int transB, float alpha, int act)
{
  A  += (long long)blockIdx.z * sA;
  Bm += (long long)blockIdx.z * sB;
  C  += (long long)blockIdx.z * sC;
  const int mBase = blockIdx.y * 64;
  const int nBase = blockIdx.x * 64;
  __shared__ _Float16 Ah[64][40];    // [m][k] f16, k contiguous (+pad)
  __shared__ _Float16 Bh[64][40];    // [n][k] f16, k contiguous (+pad)
  __shared__ float    stgA[2048];    // f32 staging for async path  [m*32+k]
  __shared__ float    stgB[2048];    // f32 staging: transB [n*32+k] else [k*64+n]
  const int tid  = threadIdx.x;
  const int wave = tid >> 5;
  const int lane = tid & 31;
  const int hl   = lane >> 4;        // lane half (K-interleave select)
  const int l16  = lane & 15;
  const int wr = wave >> 1, wc = wave & 1;   // wave -> 32x32 quadrant
  const bool alnA = ((lda & 3) == 0) && ((((unsigned long long)A) & 15) == 0);
  const bool alnB = ((ldb & 3) == 0) && ((((unsigned long long)Bm) & 15) == 0);

  v8f z = {};
  v8f acc00 = z, acc01 = z, acc10 = z, acc11 = z;

  for (int k0 = 0; k0 < K; k0 += 32) {
    const bool fullK = (k0 + 32 <= K);
    const bool fastA = fullK && alnA && (mBase + 64 <= M);
    const bool fastB = fullK && alnB && (nBase + 64 <= N);

    if (fastA) {
#pragma unroll
      for (int j = 0; j < 4; ++j) {                 // 512 16B chunks
        int c = tid + j * 128;
        int r = c >> 3, k4 = (c & 7) << 2;
        const float* ga = A + (long long)(mBase + r) * lda + (k0 + k4);
        async_g2l_b128((unsigned)(unsigned long long)&stgA[r * 32 + k4], ga);
      }
    } else {
      for (int i = tid; i < 2048; i += 128) {
        int r = i >> 5, k = i & 31;
        int gm = mBase + r, gk = k0 + k;
        Ah[r][k] = (_Float16)((gm < M && gk < K) ? A[(long long)gm * lda + gk] : 0.f);
      }
    }
    if (fastB) {
      if (transB) {
#pragma unroll
        for (int j = 0; j < 4; ++j) {
          int c = tid + j * 128;
          int r = c >> 3, k4 = (c & 7) << 2;
          const float* ga = Bm + (long long)(nBase + r) * ldb + (k0 + k4);
          async_g2l_b128((unsigned)(unsigned long long)&stgB[r * 32 + k4], ga);
        }
      } else {
#pragma unroll
        for (int j = 0; j < 4; ++j) {               // [k][n] 16 chunks of 4 cols per k
          int c = tid + j * 128;
          int k = c >> 4, n4 = (c & 15) << 2;
          const float* ga = Bm + (long long)(k0 + k) * ldb + (nBase + n4);
          async_g2l_b128((unsigned)(unsigned long long)&stgB[k * 64 + n4], ga);
        }
      }
    } else {
      for (int i = tid; i < 2048; i += 128) {
        int r = i >> 5, k = i & 31;
        int gn = nBase + r, gk = k0 + k;
        float vb;
        if (transB) vb = (gn < N && gk < K) ? Bm[(long long)gn * ldb + gk] : 0.f;
        else        vb = (gn < N && gk < K) ? Bm[(long long)gk * ldb + gn] : 0.f;
        Bh[r][k] = (_Float16)vb;
      }
    }
    if (fastA || fastB) wait_asynccnt0();
    __syncthreads();
    // f32 staging -> f16 WMMA tiles
    if (fastA) {
      for (int i = tid; i < 2048; i += 128) { int r = i >> 5, k = i & 31; Ah[r][k] = (_Float16)stgA[r * 32 + k]; }
    }
    if (fastB) {
      if (transB) { for (int i = tid; i < 2048; i += 128) { int n = i >> 5, k = i & 31; Bh[n][k] = (_Float16)stgB[n * 32 + k]; } }
      else        { for (int i = tid; i < 2048; i += 128) { int n = i >> 5, k = i & 31; Bh[n][k] = (_Float16)stgB[k * 64 + n]; } }
    }
    __syncthreads();

    // Build fragments per ISA 16-bit layout:
    // lanes 0-15: K = {0..7,16..23}; lanes 16-31: K = {8..15,24..31}; pairs per VGPR.
    auto frag = [&](_Float16 (*Tt)[40], int base) -> v16h {
      v16h f;
#pragma unroll
      for (int v = 0; v < 8; ++v) {
        int kb = (v < 4 ? 2 * v : 16 + 2 * (v - 4)) + hl * 8;
        f[2 * v]     = Tt[base + l16][kb];
        f[2 * v + 1] = Tt[base + l16][kb + 1];
      }
      return f;
    };
    v16h af0 = frag(Ah, wr * 32);
    v16h af1 = frag(Ah, wr * 32 + 16);
    v16h bf0 = frag(Bh, wc * 32);
    v16h bf1 = frag(Bh, wc * 32 + 16);
    acc00 = __builtin_amdgcn_wmma_f32_16x16x32_f16(false, af0, false, bf0, (short)0, acc00, false, false);
    acc01 = __builtin_amdgcn_wmma_f32_16x16x32_f16(false, af0, false, bf1, (short)0, acc01, false, false);
    acc10 = __builtin_amdgcn_wmma_f32_16x16x32_f16(false, af1, false, bf0, (short)0, acc10, false, false);
    acc11 = __builtin_amdgcn_wmma_f32_16x16x32_f16(false, af1, false, bf1, (short)0, acc11, false, false);
    __syncthreads();
  }

  auto store_tile = [&](const v8f& ac, int mOff, int nOff) {
    int col = nBase + nOff + l16;
    if (col < N) {
      float bv = bias ? bias[col] : 0.f;
#pragma unroll
      for (int r = 0; r < 8; ++r) {
        int row = mBase + mOff + hl * 8 + r;    // C/D layout: VGPR r -> M=r / M=8+r
        if (row < M) {
          float xv = alpha * ac[r] + bv;
          if (act == 1)      xv = fmaxf(xv, 0.f);
          else if (act == 2) xv = (xv > 0.f) ? xv : 0.01f * xv;
          else if (act == 3) xv = tanhf(xv);
          C[(long long)row * ldc + col] = xv;
        }
      }
    }
  };
  store_tile(acc00, wr * 32,      wc * 32);
  store_tile(acc01, wr * 32,      wc * 32 + 16);
  store_tile(acc10, wr * 32 + 16, wc * 32);
  store_tile(acc11, wr * 32 + 16, wc * 32 + 16);
}

// ---------------------------------------------------------------------------
// Row softmax with optional column mask (mask<=0 -> NEG before softmax).
// ---------------------------------------------------------------------------
__global__ __launch_bounds__(256)
void softmax_rows(float* __restrict__ S, const float* __restrict__ mask,
                  int cols, int ldS, int rowsPerB, int maskStride)
{
  const long long row = blockIdx.x;
  float* r = S + row * (long long)ldS;
  const float* mrow = mask ? (mask + (long long)(row / rowsPerB) * maskStride) : nullptr;
  __shared__ float red[256];
  const int tid = threadIdx.x;
  float mx = -3.0e38f;
  for (int c = tid; c < cols; c += 256) {
    float v = r[c];
    if (mrow && !(mrow[c] > 0.f)) v = kNEG;
    mx = fmaxf(mx, v);
  }
  red[tid] = mx; __syncthreads();
  for (int s = 128; s > 0; s >>= 1) { if (tid < s) red[tid] = fmaxf(red[tid], red[tid + s]); __syncthreads(); }
  mx = red[0]; __syncthreads();
  float sum = 0.f;
  for (int c = tid; c < cols; c += 256) {
    float v = r[c];
    if (mrow && !(mrow[c] > 0.f)) v = kNEG;
    sum += __expf(v - mx);
  }
  red[tid] = sum; __syncthreads();
  for (int s = 128; s > 0; s >>= 1) { if (tid < s) red[tid] += red[tid + s]; __syncthreads(); }
  float inv = 1.f / red[0];
  for (int c = tid; c < cols; c += 256) {
    float v = r[c];
    if (mrow && !(mrow[c] > 0.f)) v = kNEG;
    r[c] = __expf(v - mx) * inv;
  }
}

// x = LN(x + t) * g + b, one block per row
__global__ __launch_bounds__(256)
void ln_residual(float* __restrict__ x, const float* __restrict__ t,
                 const float* __restrict__ g, const float* __restrict__ b, int D)
{
  const long long row = blockIdx.x;
  float* xr = x + row * D;
  const float* tr = t + row * D;
  __shared__ float r1[256], r2[256];
  const int tid = threadIdx.x;
  float s = 0.f, s2 = 0.f;
  for (int i = tid; i < D; i += 256) { float v = xr[i] + tr[i]; s += v; s2 += v * v; }
  r1[tid] = s; r2[tid] = s2; __syncthreads();
  for (int st = 128; st > 0; st >>= 1) { if (tid < st) { r1[tid] += r1[tid + st]; r2[tid] += r2[tid + st]; } __syncthreads(); }
  float mu = r1[0] / D;
  float var = r2[0] / D - mu * mu;
  float inv = rsqrtf(var + 1e-5f);
  for (int i = tid; i < D; i += 256) { float v = xr[i] + tr[i]; xr[i] = (v - mu) * inv * g[i] + b[i]; }
}

// ---------------------------------------------------------------------------
// Elementwise / fused small kernels
// ---------------------------------------------------------------------------
__global__ void k_fill(float* p, float v, long long n)
{ long long i = blockIdx.x * 256LL + threadIdx.x; if (i < n) p[i] = v; }

// E = ce + ne = code_x*c_emb + neighbors*n_emb   (B,T,N,48)
__global__ void k_embed_sum(const float* cx, const float* nb, const float* ce, const float* ne,
                            float* E, long long n)
{
  long long i = blockIdx.x * 256LL + threadIdx.x; if (i >= n) return;
  int c = (int)(i % kCODE);
  long long t = i / kCODE;
  int node = (int)(t % kN);
  long long bt = t / kN;
  E[i] = cx[bt * kN + node] * ce[node * kCODE + c] + nb[bt * kN + node] * ne[node * kCODE + c];
}

// Aco = ce + code_x*S ; Ano = ne + neighbors*S   (S = adj@(ce+ne))
__global__ void k_graph_pre(const float* cx, const float* nb, const float* ce, const float* ne,
                            const float* S, float* Aco, float* Ano, long long n)
{
  long long i = blockIdx.x * 256LL + threadIdx.x; if (i >= n) return;
  int c = (int)(i % kCODE);
  long long t = i / kCODE;
  int node = (int)(t % kN);
  long long bt = t / kN;
  float xv = cx[bt * kN + node], nv = nb[bt * kN + node], sv = S[i];
  Aco[i] = xv * ce[node * kCODE + c] + xv * sv;
  Ano[i] = nv * ne[node * kCODE + c] + nv * sv;
}

// GRU gate fusion -> h_m1
__global__ void k_gru_fuse(const float* gi, const float* gh, const float* h, float* hm1, long long n)
{
  long long i = blockIdx.x * 256LL + threadIdx.x; if (i >= n) return;
  int d = (int)(i % kH);
  long long bn = i / kH;
  long long gb = bn * (3 * kH);
  float ir = gi[gb + d], iz = gi[gb + kH + d], in = gi[gb + 2 * kH + d];
  float hr = gh[gb + d], hz = gh[gb + kH + d], hn = gh[gb + 2 * kH + d];
  float r  = sigmoidf_(ir + hr);
  float z  = sigmoidf_(iz + hz);
  float nn = tanhf(in + r * hn);
  hm1[i] = (1.f - z) * nn + z * h[i];
}

// q = m2*prev_no + m3*u_emb ; also write m23 mask (B,N)
__global__ void k_qmask(const float* divided, const float* prevno, const float* uemb,
                        float* q, float* mask1, int t, long long n)
{
  long long i = blockIdx.x * 256LL + threadIdx.x; if (i >= n) return;  // n = B*N*32
  int g = (int)(i % kG);
  long long bn = i / kG;
  int node = (int)(bn % kN);
  int b = (int)(bn / kN);
  const float* dv = divided + (((long long)(b * kT + t) * kN + node) * 3);
  float m2 = dv[1], m3 = dv[2];
  q[i] = m2 * prevno[i] + m3 * uemb[node * kG + g];
  if (g == 0) mask1[bn] = m2 + m3;
}

// Per-batch masked max pooling of hm1/hm23 -> outs[b,t,:]; also stores m23sum.
__global__ __launch_bounds__(256)
void k_scan_out(const float* divided, const float* hm1, const float* hm23,
                float* outs, float* scal, int t)
{
  const int b = blockIdx.x, tid = threadIdx.x;
  __shared__ float r1[256], r2[256];
  float s1 = 0.f, s23 = 0.f;
  for (int n0 = tid; n0 < kN; n0 += 256) {
    const float* dv = divided + (((long long)(b * kT + t) * kN + n0) * 3);
    s1 += dv[0]; s23 += dv[1] + dv[2];
  }
  r1[tid] = s1; r2[tid] = s23; __syncthreads();
  for (int s = 128; s > 0; s >>= 1) { if (tid < s) { r1[tid] += r1[tid + s]; r2[tid] += r2[tid + s]; } __syncthreads(); }
  float m1sum = r1[0], m23sum = r2[0];
  if (tid == 0) scal[b] = m23sum;
  bool use23 = (t > 0) && (m23sum > 0.f);
  if (tid < kH) {
    float mx1 = kNEG, mx23 = kNEG;
    for (int n0 = 0; n0 < kN; ++n0) {
      const float* dv = divided + (((long long)(b * kT + t) * kN + n0) * 3);
      float m1 = dv[0], m23 = dv[1] + dv[2];
      long long idx = ((long long)b * kN + n0) * kH + tid;
      if (m1 > 0.f)  mx1  = fmaxf(mx1,  hm1[idx]);
      if (m23 > 0.f) mx23 = fmaxf(mx23, hm23[idx]);
    }
    float o1  = (m1sum > 0.f) ? mx1  : 0.f;
    float o23 = use23         ? mx23 : 0.f;
    outs[((long long)b * kT + t) * kH + tid] = o1 + o23;
  }
}

__global__ void k_h_update(const float* divided, const float* hm1, const float* hm23,
                           const float* scal, const int* lens, float* h, int t, long long n)
{
  long long i = blockIdx.x * 256LL + threadIdx.x; if (i >= n) return;  // n = B*N*150
  long long bn = i / kH;
  int node = (int)(bn % kN);
  int b = (int)(bn / kN);
  const float* dv = divided + (((long long)(b * kT + t) * kN + node) * 3);
  float m1 = dv[0], m23 = dv[1] + dv[2];
  bool use23 = (t > 0) && (scal[b] > 0.f);
  float hnew = m1 * hm1[i] + (use23 ? m23 * hm23[i] : 0.f);
  if (t < lens[b]) h[i] = hnew;
}

__global__ void k_prevno_update(const float* no, const int* lens, float* prevno, int t, long long n)
{
  long long i = blockIdx.x * 256LL + threadIdx.x; if (i >= n) return;  // n = B*N*32
  int g = (int)(i % kG);
  long long bn = i / kG;
  int node = (int)(bn % kN);
  int b = (int)(bn / kN);
  if (t < lens[b]) prevno[i] = no[(((long long)(b * kT + t) * kN + node) * kG) + g];
}

// attention-pool over T -> icd[b,:]
__global__ __launch_bounds__(256)
void k_pool(const float* outs, const float* pW, const float* pb, const float* pctx,
            const int* lens, float* icd)
{
  const int b = blockIdx.x, tid = threadIdx.x;
  const int t = tid >> 5, lane = tid & 31;   // 8 waves = 8 timesteps, 32 pool cols
  __shared__ float scs[kT];
  __shared__ float wts[kT];
  float part = pb[lane];
  const float* orow = outs + ((long long)b * kT + t) * kH;
  for (int k = 0; k < kH; ++k) part += orow[k] * pW[k * 32 + lane];
  float val = part * pctx[lane];
  for (int m = 16; m > 0; m >>= 1) val += __shfl_xor(val, m, 32);
  if (lane == 0) scs[t] = val;
  __syncthreads();
  if (tid == 0) {
    int len = lens[b];
    float mx = -3e38f;
    for (int tt = 0; tt < kT; ++tt) { float v = (tt < len) ? scs[tt] : kNEG; if (v > mx) mx = v; }
    float s = 0.f;
    for (int tt = 0; tt < kT; ++tt) { float v = (tt < len) ? scs[tt] : kNEG; wts[tt] = __expf(v - mx); s += wts[tt]; }
    for (int tt = 0; tt < kT; ++tt) wts[tt] /= s;
  }
  __syncthreads();
  if (tid < kH) {
    float acc = 0.f;
    for (int tt = 0; tt < kT; ++tt) acc += wts[tt] * outs[((long long)b * kT + tt) * kH + tid];
    icd[b * kH + tid] = acc;
  }
}

__global__ void k_gather(const int* note, const float* temb, float* xe, long long n)
{
  long long i = blockIdx.x * 256LL + threadIdx.x; if (i >= n) return;
  int c = (int)(i % kEMB);
  long long bl = i / kEMB;
  xe[i] = temb[(long long)note[bl] * kEMB + c];
}

__global__ void k_add_pe(float* x, long long n)
{
  long long i = blockIdx.x * 256LL + threadIdx.x; if (i >= n) return;
  int d = (int)(i % kD);
  int l = (int)((i / kD) % kL);
  float expo = (float)(2 * (d / 2)) / (float)kD;
  float ang = (float)l * __powf(10000.f, -expo);
  x[i] += (d & 1) ? __cosf(ang) : __sinf(ang);
}

// e[b,l] = tanh(qp[b]+kp[b,l]) . xa_v + xa_b (masked)
__global__ void k_xatt_e(const float* qp, const float* kp, const float* xav, const float* xab,
                         const float* nmask, float* e, long long n)
{
  long long i = blockIdx.x * 256LL + threadIdx.x; if (i >= n) return;  // n = B*L
  int b = (int)(i / kL);
  float s = 0.f;
  for (int j = 0; j < kXA; ++j) s += tanhf(qp[b * kXA + j] + kp[i * kXA + j]) * xav[j];
  s += xab[0];
  e[i] = (nmask[i] > 0.f) ? s : kNEG;
}

// ctx[b,:] = softmax(e[b,:]) @ hid[b]
__global__ __launch_bounds__(256)
void k_xatt_ctx(const float* e, const float* x, float* ctx)
{
  const int b = blockIdx.x, tid = threadIdx.x;
  __shared__ float w[kL];
  __shared__ float red[256];
  float v = (tid < kL) ? e[b * kL + tid] : -3e38f;
  red[tid] = v; __syncthreads();
  for (int s = 128; s > 0; s >>= 1) { if (tid < s) red[tid] = fmaxf(red[tid], red[tid + s]); __syncthreads(); }
  float mx = red[0]; __syncthreads();
  float ex = (tid < kL) ? __expf(e[b * kL + tid] - mx) : 0.f;
  red[tid] = ex; __syncthreads();
  for (int s = 128; s > 0; s >>= 1) { if (tid < s) red[tid] += red[tid + s]; __syncthreads(); }
  float inv = 1.f / red[0];
  if (tid < kL) w[tid] = ex * inv;
  __syncthreads();
  for (int d = tid; d < kD; d += 256) {
    float acc = 0.f;
    for (int l = 0; l < kL; ++l) acc += w[l] * x[((long long)b * kL + l) * kD + d];
    ctx[b * kD + d] = acc;
  }
}

__global__ __launch_bounds__(256)
void k_final(const float* icd, const float* ctx, const float* cW, const float* cb, float* out)
{
  const int b = blockIdx.x, tid = threadIdx.x;
  __shared__ float red[256];
  float s = 0.f;
  for (int i = tid; i < kH + kD; i += 256) {
    float f = (i < kH) ? icd[b * kH + i] : ctx[b * kD + (i - kH)];
    s += f * cW[i];
  }
  red[tid] = s; __syncthreads();
  for (int st = 128; st > 0; st >>= 1) { if (tid < st) red[tid] += red[tid + st]; __syncthreads(); }
  if (tid == 0) out[b] = 1.f / (1.f + __expf(-(red[0] + cb[0])));
}

// ---------------------------------------------------------------------------
extern "C" void kernel_launch(void* const* d_in, const int* in_sizes, int n_in,
                              void* d_out, int out_size, void* d_ws, size_t ws_size,
                              hipStream_t stream)
{
  (void)in_sizes; (void)n_in; (void)out_size; (void)ws_size;
  const float* code_x    = (const float*)d_in[0];
  const float* divided   = (const float*)d_in[1];
  const float* neighbors = (const float*)d_in[2];
  const int*   lens      = (const int*)d_in[3];
  const int*   note      = (const int*)d_in[4];
  const float* note_mask = (const float*)d_in[5];
  const float* adj       = (const float*)d_in[6];
  const float* c_emb = (const float*)d_in[7],  *n_emb = (const float*)d_in[8], *u_emb = (const float*)d_in[9];
  const float* Wg = (const float*)d_in[10], *bg = (const float*)d_in[11];
  const float* gWi = (const float*)d_in[12], *gWh = (const float*)d_in[13];
  const float* gbi = (const float*)d_in[14], *gbh = (const float*)d_in[15];
  const float* taWq = (const float*)d_in[16], *tabq = (const float*)d_in[17];
  const float* taWk = (const float*)d_in[18], *tabk = (const float*)d_in[19];
  const float* taWv = (const float*)d_in[20], *tabv = (const float*)d_in[21];
  const float* poolW = (const float*)d_in[22], *poolb = (const float*)d_in[23], *poolctx = (const float*)d_in[24];
  const float* tok_emb = (const float*)d_in[25], *projW = (const float*)d_in[26], *projb = (const float*)d_in[27];
  const float* xaWq = (const float*)d_in[60], *xabq = (const float*)d_in[61];
  const float* xaWk = (const float*)d_in[62], *xabk = (const float*)d_in[63];
  const float* xav  = (const float*)d_in[64], *xab  = (const float*)d_in[65];
  const float* clsW = (const float*)d_in[66], *clsb = (const float*)d_in[67];
  float* out = (float*)d_out;

  // ---- workspace arena (floats) ----
  float* wsf = (float*)d_ws;
  size_t cur = 0;
  auto take = [&](size_t nf) -> float* { float* p = wsf + cur; cur += (nf + 63) & ~(size_t)63; return p; };

  float* bigA = take(18000000);               // 72MB time-shared region
  float* E    = bigA;                         // graph phase
  float* Sg   = bigA + 4608000;
  float* Aco  = bigA + 9216000;
  float* sbuf = bigA;                         // scan phase: (B,1500,1500) scores
  float* xe     = bigA;                       // text phase
  float* x      = bigA + 131072;
  float* xq     = bigA + 917504;
  float* xk     = bigA + 1703936;
  float* xv     = bigA + 2490368;
  float* xo     = bigA + 3276800;
  float* tmp    = bigA + 4063232;
  float* scores = bigA + 4849664;
  float* ffn    = bigA + 5373952;
  float* kp     = bigA + 8519680;
  float* qp     = bigA + 8585216;
  float* ebuf   = bigA + 8585728;
  float* ctx    = bigA + 8586752;

  float* co    = take(3072000);
  float* no    = take(3072000);
  float* h     = take(1800000);
  float* prevn = take(384000);
  float* gi    = take(5400000);
  float* gh    = take(5400000);
  float* hm1   = take(1800000);
  float* qb    = take(384000);
  float* qh    = take(384000);
  float* kh    = take(384000);
  float* vv    = take(1800000);
  float* hm23  = take(1800000);
  float* mask1 = take(12032);
  float* scal  = take(64);
  float* outs  = take(9600);
  float* icd   = take(1280);

  auto gemm = [&](const float* Ap, const float* Bp, const float* biasp, float* Cp,
                  int M, int Nc, int K, int lda, int ldb, int ldc,
                  long long sA, long long sB, long long sC, int batch,
                  int transB, float alpha, int act) {
    dim3 g((Nc + 63) / 64, (M + 63) / 64, batch);
    gemm_wmma<<<g, 128, 0, stream>>>(Ap, Bp, biasp, Cp, M, Nc, K, lda, ldb, ldc, sA, sB, sC, transB, alpha, act);
  };
  auto ew = [&](long long n) { return dim3((unsigned)((n + 255) / 256)); };

  // ---------------- graph phase ----------------
  const long long nE = (long long)kB * kT * kN * kCODE;
  k_embed_sum<<<ew(nE), 256, 0, stream>>>(code_x, neighbors, c_emb, n_emb, E, nE);
  // S = adj @ (ce+ne), batched over (b,t)
  gemm(adj, E, nullptr, Sg, kN, kCODE, kN, kN, kCODE, kCODE,
       0, (long long)kN * kCODE, (long long)kN * kCODE, kB * kT, 0, 1.f, 0);
  k_graph_pre<<<ew(nE), 256, 0, stream>>>(code_x, neighbors, c_emb, n_emb, Sg, Aco, E, nE);
  gemm(Aco, Wg, bg, co, kB * kT * kN, kG, kCODE, kCODE, kG, kG, 0, 0, 0, 1, 0, 1.f, 2);  // lrelu
  gemm(E,   Wg, bg, no, kB * kT * kN, kG, kCODE, kCODE, kG, kG, 0, 0, 0, 1, 0, 1.f, 2);

  // ---------------- scan over T ----------------
  k_fill<<<ew(1800000), 256, 0, stream>>>(h, 0.f, 1800000);
  k_fill<<<ew(384000), 256, 0, stream>>>(prevn, 0.f, 384000);
  const float rsqTA = 0.17677669529663687f;   // 1/sqrt(32)
  for (int t = 0; t < kT; ++t) {
    gemm(co + (long long)t * kN * kG, gWi, gbi, gi, kN, 3 * kH, kG, kG, 3 * kH, 3 * kH,
         (long long)kT * kN * kG, 0, (long long)kN * 3 * kH, kB, 0, 1.f, 0);
    gemm(h, gWh, gbh, gh, kN, 3 * kH, kH, kH, 3 * kH, 3 * kH,
         (long long)kN * kH, 0, (long long)kN * 3 * kH, kB, 0, 1.f, 0);
    const long long nH = (long long)kB * kN * kH;
    k_gru_fuse<<<ew(nH), 256, 0, stream>>>(gi, gh, h, hm1, nH);
    const long long nG = (long long)kB * kN * kG;
    k_qmask<<<ew(nG), 256, 0, stream>>>(divided, prevn, u_emb, qb, mask1, t, nG);
    gemm(qb, taWq, tabq, qh, kN, 32, kG, kG, 32, 32, (long long)kN * kG, 0, (long long)kN * 32, kB, 0, 1.f, 0);
    gemm(qb, taWk, tabk, kh, kN, 32, kG, kG, 32, 32, (long long)kN * kG, 0, (long long)kN * 32, kB, 0, 1.f, 0);
    gemm(co + (long long)t * kN * kG, taWv, tabv, vv, kN, kH, kG, kG, kH, kH,
         (long long)kT * kN * kG, 0, (long long)kN * kH, kB, 0, 1.f, 0);
    gemm(qh, kh, nullptr, sbuf, kN, kN, 32, 32, 32, kN,
         (long long)kN * 32, (long long)kN * 32, (long long)kN * kN, kB, 1, rsqTA, 0);
    softmax_rows<<<kB * kN, 256, 0, stream>>>(sbuf, mask1, kN, kN, kN, kN);
    gemm(sbuf, vv, nullptr, hm23, kN, kH, kN, kN, kH, kH,
         (long long)kN * kN, (long long)kN * kH, (long long)kN * kH, kB, 0, 1.f, 3);  // tanh
    k_scan_out<<<kB, 256, 0, stream>>>(divided, hm1, hm23, outs, scal, t);
    k_h_update<<<ew(nH), 256, 0, stream>>>(divided, hm1, hm23, scal, lens, h, t, nH);
    k_prevno_update<<<ew(nG), 256, 0, stream>>>(no, lens, prevn, t, nG);
  }
  k_pool<<<kB, 256, 0, stream>>>(outs, poolW, poolb, poolctx, lens, icd);

  // ---------------- text encoder ----------------
  const long long nXE = (long long)kB * kL * kEMB;
  k_gather<<<ew(nXE), 256, 0, stream>>>(note, tok_emb, xe, nXE);
  gemm(xe, projW, projb, x, kB * kL, kD, kEMB, kEMB, kD, kD, 0, 0, 0, 1, 0, 1.f, 0);
  const long long nX = (long long)kB * kL * kD;
  k_add_pe<<<ew(nX), 256, 0, stream>>>(x, nX);
  const float rsqHD = 0.07216878364870323f;   // 1/sqrt(192)
  for (int l = 0; l < 2; ++l) {
    const int base = 28 + 16 * l;
    const float* Wq = (const float*)d_in[base + 0],  *bq  = (const float*)d_in[base + 1];
    const float* Wk = (const float*)d_in[base + 2],  *bk  = (const float*)d_in[base + 3];
    const float* Wv = (const float*)d_in[base + 4],  *bv  = (const float*)d_in[base + 5];
    const float* Wo = (const float*)d_in[base + 6],  *bo  = (const float*)d_in[base + 7];
    const float* g1 = (const float*)d_in[base + 8],  *b1g = (const float*)d_in[base + 9];
    const float* g2 = (const float*)d_in[base + 10], *b2g = (const float*)d_in[base + 11];
    const float* W1 = (const float*)d_in[base + 12], *bb1 = (const float*)d_in[base + 13];
    const float* W2 = (const float*)d_in[base + 14], *bb2 = (const float*)d_in[base + 15];
    gemm(x, Wq, bq, xq, kB * kL, kD, kD, kD, kD, kD, 0, 0, 0, 1, 0, 1.f, 0);
    gemm(x, Wk, bk, xk, kB * kL, kD, kD, kD, kD, kD, 0, 0, 0, 1, 0, 1.f, 0);
    gemm(x, Wv, bv, xv, kB * kL, kD, kD, kD, kD, kD, 0, 0, 0, 1, 0, 1.f, 0);
    for (int hh = 0; hh < kHEADS; ++hh) {
      gemm(xq + hh * kHD, xk + hh * kHD, nullptr, scores + (long long)hh * kL * kL,
           kL, kL, kHD, kD, kD, kL,
           (long long)kL * kD, (long long)kL * kD, (long long)kHEADS * kL * kL, kB, 1, rsqHD, 0);
    }
    softmax_rows<<<kB * kHEADS * kL, 256, 0, stream>>>(scores, note_mask, kL, kL, kHEADS * kL, kL);
    for (int hh = 0; hh < kHEADS; ++hh) {
      gemm(scores + (long long)hh * kL * kL, xv + hh * kHD, nullptr, xo + hh * kHD,
           kL, kHD, kL, kL, kD, kD,
           (long long)kHEADS * kL * kL, (long long)kL * kD, (long long)kL * kD, kB, 0, 1.f, 0);
    }
    gemm(xo, Wo, bo, tmp, kB * kL, kD, kD, kD, kD, kD, 0, 0, 0, 1, 0, 1.f, 0);
    ln_residual<<<kB * kL, 256, 0, stream>>>(x, tmp, g1, b1g, kD);
    gemm(x, W1, bb1, ffn, kB * kL, kFF, kD, kD, kFF, kFF, 0, 0, 0, 1, 0, 1.f, 1);     // relu
    gemm(ffn, W2, bb2, tmp, kB * kL, kD, kFF, kFF, kD, kD, 0, 0, 0, 1, 0, 1.f, 0);
    ln_residual<<<kB * kL, 256, 0, stream>>>(x, tmp, g2, b2g, kD);
  }

  // ---------------- cross attention + classifier ----------------
  gemm(x, xaWk, xabk, kp, kB * kL, kXA, kD, kD, kXA, kXA, 0, 0, 0, 1, 0, 1.f, 0);
  gemm(icd, xaWq, xabq, qp, kB, kXA, kH, kH, kXA, kXA, 0, 0, 0, 1, 0, 1.f, 0);
  const long long nBL = (long long)kB * kL;
  k_xatt_e<<<ew(nBL), 256, 0, stream>>>(qp, kp, xav, xab, note_mask, ebuf, nBL);
  k_xatt_ctx<<<kB, 256, 0, stream>>>(ebuf, x, ctx);
  k_final<<<kB, 256, 0, stream>>>(icd, ctx, clsW, clsb, out);
}
